// CCA2_18786186953483
// MI455X (gfx1250) — matrix operand order repaired and checked by reading, not compiled, not run on hardware
//
#include <hip/hip_runtime.h>
#include <math.h>

typedef __attribute__((ext_vector_type(16))) _Float16 v16h;
typedef __attribute__((ext_vector_type(8)))  _Float16 v8h;
typedef __attribute__((ext_vector_type(8)))  float    v8f;
typedef int v4i_vs __attribute__((vector_size(16)));

#define HEADS 8

#if defined(__HIP_DEVICE_COMPILE__) && \
    __has_builtin(__builtin_amdgcn_global_load_async_to_lds_b128)
#define HAVE_ASYNC 1
#else
#define HAVE_ASYNC 0
#endif

// ---------------------------------------------------------------------------
// helpers
// ---------------------------------------------------------------------------
__device__ __forceinline__ v16h combine8(v8h lo, v8h hi) {
  v16h r;
#pragma unroll
  for (int i = 0; i < 8; ++i) { r[i] = lo[i]; r[i + 8] = hi[i]; }
  return r;
}

__device__ __forceinline__ v8f wmma_f16(v16h a, v16h b, v8f c) {
  return __builtin_amdgcn_wmma_f32_16x16x32_f16(
      /*neg_a=*/false, a, /*neg_b=*/false, b,
      /*c_mod=*/(short)0, c, /*reuse_a=*/false, /*reuse_b=*/false);
}

// async 16-byte global -> LDS copy (ASYNCcnt-tracked on gfx1250)
__device__ __forceinline__ void async_copy16(const _Float16* g, _Float16* l) {
#if HAVE_ASYNC
  typedef __attribute__((address_space(1))) v4i_vs* gptr_t;
  typedef __attribute__((address_space(3))) v4i_vs* lptr_t;
  __builtin_amdgcn_global_load_async_to_lds_b128(
      (gptr_t)(void*)(const void*)g, (lptr_t)(void*)l, 0, 0);
#else
  *(v8h*)l = *(const v8h*)g;
#endif
}

__device__ __forceinline__ void async_wait_all() {
#if HAVE_ASYNC
  asm volatile("s_wait_asynccnt 0x0" ::: "memory");
#endif
}

// ---------------------------------------------------------------------------
// f32 -> f16 converts
// ---------------------------------------------------------------------------
__global__ void cvt_f32_f16(const float* __restrict__ in,
                            _Float16* __restrict__ out, int n) {
  int i = blockIdx.x * blockDim.x + threadIdx.x;
  if (i < n) out[i] = (_Float16)in[i];
}

// x [8192,512] f32 -> cat[:, 0:512] (row stride 1536) f16
__global__ void cvt_x_to_cat(const float* __restrict__ x,
                             _Float16* __restrict__ cat, int n) {
  int i = blockIdx.x * blockDim.x + threadIdx.x;
  if (i < n) {
    int row = i >> 9;
    int col = i & 511;
    cat[(size_t)row * 1536 + col] = (_Float16)x[i];
  }
}

// ---------------------------------------------------------------------------
// Generic f16 WMMA GEMM:  Out = (A[M,K](lda) x W[K,N] + bias) * scale
// block = 256 threads (8 waves), block tile 128(M) x 128(N), K step = 32,
// double-buffered LDS with async global->LDS staging for A.
// Wave (wr,wc) computes a 32x64 tile = 2x4 WMMA 16x16 tiles.
// ---------------------------------------------------------------------------
template <bool OUT_F16>
__global__ __launch_bounds__(256) void gemm_wmma(
    const _Float16* __restrict__ A, const _Float16* __restrict__ W,
    const float* __restrict__ bias, void* __restrict__ Out,
    int M, int N, int K, int lda, float scale) {
  __shared__ __align__(16) _Float16 sA[2][128 * 32];   // [m][k]
  __shared__ __align__(16) _Float16 sB[2][128 * 32];   // [n][k] (transposed)

  const int tid  = threadIdx.x;
  const int wid  = tid >> 5;
  const int lane = tid & 31;
  const int l16  = lane & 15;
  const int half = lane >> 4;
  const int wr   = wid >> 1;   // 0..3  (M)
  const int wc   = wid & 1;    // 0..1  (N)
  const int bm   = blockIdx.y * 128;
  const int bn   = blockIdx.x * 128;

  // ---- staging lambdas ----
  auto stageA = [&](int buf, int k0) {
#pragma unroll
    for (int j = 0; j < 2; ++j) {
      int c    = tid + j * 256;        // 0..511
      int row  = c >> 2;
      int col8 = (c & 3) * 8;
      async_copy16(&A[(size_t)(bm + row) * lda + k0 + col8],
                   &sA[buf][row * 32 + col8]);
    }
  };
  auto stageB = [&](int buf, int k0) {
#pragma unroll
    for (int j = 0; j < 2; ++j) {
      int c  = tid + j * 256;          // 0..511
      int k  = c >> 4;                 // 0..31
      int n8 = (c & 15) * 8;           // 0..120
      v8h w = *(const v8h*)&W[(size_t)(k0 + k) * N + bn + n8];
#pragma unroll
      for (int t = 0; t < 8; ++t) sB[buf][(n8 + t) * 32 + k] = w[t];
    }
  };

  v8f acc[2][4];
#pragma unroll
  for (int tm = 0; tm < 2; ++tm)
#pragma unroll
    for (int tn = 0; tn < 4; ++tn) acc[tm][tn] = (v8f)0.0f;

  stageA(0, 0);
  stageB(0, 0);
  int buf = 0;
  for (int k0 = 0; k0 < K; k0 += 32) {
    async_wait_all();
    __syncthreads();
    if (k0 + 32 < K) {                 // overlap next tile with compute
      stageA(buf ^ 1, k0 + 32);
      stageB(buf ^ 1, k0 + 32);
    }

    // ---- fragments (all preloaded, then WMMA chain) ----
    v16h afrag[2], bfrag[4];
#pragma unroll
    for (int tm = 0; tm < 2; ++tm) {
      int row = wr * 32 + tm * 16 + l16;
      const _Float16* p = &sA[buf][row * 32 + half * 8];
      afrag[tm] = combine8(*(const v8h*)p, *(const v8h*)(p + 16));
    }
#pragma unroll
    for (int tn = 0; tn < 4; ++tn) {
      int n = wc * 64 + tn * 16 + l16;
      const _Float16* p = &sB[buf][n * 32 + half * 8];
      bfrag[tn] = combine8(*(const v8h*)p, *(const v8h*)(p + 16));
    }
#pragma unroll
    for (int tm = 0; tm < 2; ++tm)
#pragma unroll
      for (int tn = 0; tn < 4; ++tn)
        acc[tm][tn] = wmma_f16(afrag[tm], bfrag[tn], acc[tm][tn]);
    buf ^= 1;
  }

  // ---- epilogue: C/D layout row = v + 8*half (lanes 0-15 M=v, 16-31 M=v+8)
#pragma unroll
  for (int tm = 0; tm < 2; ++tm)
#pragma unroll
    for (int tn = 0; tn < 4; ++tn)
#pragma unroll
      for (int v = 0; v < 8; ++v) {
        int row = bm + wr * 32 + tm * 16 + v + half * 8;
        int col = bn + wc * 64 + tn * 16 + l16;
        float r = (acc[tm][tn][v] + bias[col]) * scale;
        if (OUT_F16)
          ((_Float16*)Out)[(size_t)row * N + col] = (_Float16)r;
        else
          ((float*)Out)[(size_t)row * N + col] = r;
      }
}

// ---------------------------------------------------------------------------
// Flash cross-attention. Q pre-scaled f16 [B*1024, 512]. KV f16 [B*S, 1024]
// (k = cols [0,512), v = cols [512,1024), head h at h*64).
// Block = 128 threads (4 waves); block handles (b, h, 64 L-rows);
// each wave owns 16 L-rows. S consumed in 64-wide chunks, double-buffered
// K/V staging (K via async global->LDS).
// ---------------------------------------------------------------------------
__global__ __launch_bounds__(128) void attn_wmma(
    const _Float16* __restrict__ Q, const _Float16* __restrict__ KV,
    _Float16* __restrict__ catOut, int S, int catOff) {
  __shared__ __align__(16) _Float16 sK[2][64 * 64];    // [s][d]
  __shared__ __align__(16) _Float16 sV[2][64 * 64];    // [d][s] (transposed)
  __shared__ __align__(16) _Float16 sP[4][16 * 64];    // per-wave P tile [m][s]

  const int tid  = threadIdx.x;
  const int wid  = tid >> 5;
  const int lane = tid & 31;
  const int l16  = lane & 15;
  const int half = lane >> 4;
  const int l0   = blockIdx.x * 64;
  const int h    = blockIdx.y;
  const int b    = blockIdx.z;

  auto stageK = [&](int buf, int s0) {
#pragma unroll
    for (int j = 0; j < 4; ++j) {
      int c  = tid + j * 128;          // 0..511
      int s  = c >> 3;
      int d8 = (c & 7) * 8;
      async_copy16(&KV[(size_t)(b * S + s0 + s) * 1024 + h * 64 + d8],
                   &sK[buf][s * 64 + d8]);
    }
  };
  auto stageV = [&](int buf, int s0) {
#pragma unroll
    for (int j = 0; j < 4; ++j) {
      int c  = tid + j * 128;          // 0..511
      int s  = c >> 3;
      int d8 = (c & 7) * 8;
      v8h w = *(const v8h*)&KV[(size_t)(b * S + s0 + s) * 1024 + 512 +
                               h * 64 + d8];
#pragma unroll
      for (int t = 0; t < 8; ++t) sV[buf][(d8 + t) * 64 + s] = w[t];
    }
  };

  // ---- Q fragments (16x64 per wave, held in registers) ----
  v16h qf[2];
  {
    const _Float16* qp =
        Q + (size_t)(b * 1024 + l0 + wid * 16 + l16) * 512 + h * 64;
#pragma unroll
    for (int kk = 0; kk < 2; ++kk) {
      const _Float16* p = qp + kk * 32 + half * 8;
      qf[kk] = combine8(*(const v8h*)p, *(const v8h*)(p + 16));
    }
  }

  float rowm[8], rowl[8];
  v8f oacc[4];
#pragma unroll
  for (int v = 0; v < 8; ++v) { rowm[v] = -1e30f; rowl[v] = 0.0f; }
#pragma unroll
  for (int tn = 0; tn < 4; ++tn) oacc[tn] = (v8f)0.0f;

  const int nch = S >> 6;
  stageK(0, 0);
  stageV(0, 0);
  int buf = 0;
  for (int ch = 0; ch < nch; ++ch) {
    async_wait_all();
    __syncthreads();
    if (ch + 1 < nch) {                // overlap next chunk with compute
      stageK(buf ^ 1, (ch + 1) * 64);
      stageV(buf ^ 1, (ch + 1) * 64);
    }

    // ---- S_ij = Q x K^T : preload all 8 B fragments, then WMMA chain ----
    v16h kf[4][2];
#pragma unroll
    for (int nt = 0; nt < 4; ++nt) {
      int sIdx = nt * 16 + l16;        // B column = s index
#pragma unroll
      for (int kk = 0; kk < 2; ++kk) {
        const _Float16* p = &sK[buf][sIdx * 64 + kk * 32 + half * 8];
        kf[nt][kk] = combine8(*(const v8h*)p, *(const v8h*)(p + 16));
      }
    }
    v8f sacc[4];
#pragma unroll
    for (int nt = 0; nt < 4; ++nt) sacc[nt] = (v8f)0.0f;
#pragma unroll
    for (int nt = 0; nt < 4; ++nt)
#pragma unroll
      for (int kk = 0; kk < 2; ++kk)
        sacc[nt] = wmma_f16(qf[kk], kf[nt][kk], sacc[nt]);

    // ---- online softmax (reductions stay inside 16-lane halves) ----
#pragma unroll
    for (int v = 0; v < 8; ++v) {
      float mx = fmaxf(fmaxf(sacc[0][v], sacc[1][v]),
                       fmaxf(sacc[2][v], sacc[3][v]));
#pragma unroll
      for (int m = 1; m < 16; m <<= 1) mx = fmaxf(mx, __shfl_xor(mx, m, 32));
      float mnew  = fmaxf(rowm[v], mx);
      float alpha = __expf(rowm[v] - mnew);
      float sum = 0.0f;
#pragma unroll
      for (int nt = 0; nt < 4; ++nt) {
        float p = __expf(sacc[nt][v] - mnew);
        sacc[nt][v] = p;
        sum += p;
      }
#pragma unroll
      for (int m = 1; m < 16; m <<= 1) sum += __shfl_xor(sum, m, 32);
      rowl[v] = rowl[v] * alpha + sum;
      rowm[v] = mnew;
#pragma unroll
      for (int tn = 0; tn < 4; ++tn) oacc[tn][v] *= alpha;
    }

    // ---- P: C-layout -> per-wave LDS tile -> A-layout ----
#pragma unroll
    for (int nt = 0; nt < 4; ++nt)
#pragma unroll
      for (int v = 0; v < 8; ++v) {
        int row = v + half * 8;
        int col = nt * 16 + l16;
        sP[wid][row * 64 + col] = (_Float16)sacc[nt][v];
      }

    // ---- O += P x V : preload all fragments, then WMMA chain ----
    v16h af[2], vf[4][2];
#pragma unroll
    for (int kk = 0; kk < 2; ++kk) {
      const _Float16* ap = &sP[wid][l16 * 64 + kk * 32 + half * 8];
      af[kk] = combine8(*(const v8h*)ap, *(const v8h*)(ap + 16));
#pragma unroll
      for (int tn = 0; tn < 4; ++tn) {
        int d = tn * 16 + l16;         // B column = head dim
        const _Float16* bp = &sV[buf][d * 64 + kk * 32 + half * 8];
        vf[tn][kk] = combine8(*(const v8h*)bp, *(const v8h*)(bp + 16));
      }
    }
#pragma unroll
    for (int tn = 0; tn < 4; ++tn)
#pragma unroll
      for (int kk = 0; kk < 2; ++kk)
        oacc[tn] = wmma_f16(af[kk], vf[tn][kk], oacc[tn]);

    buf ^= 1;
  }

  // ---- normalize + scatter into cat buffer ----
#pragma unroll
  for (int v = 0; v < 8; ++v) {
    float inv = 1.0f / rowl[v];
#pragma unroll
    for (int tn = 0; tn < 4; ++tn) {
      int row = b * 1024 + l0 + wid * 16 + v + half * 8;
      int col = catOff + h * 64 + tn * 16 + l16;
      catOut[(size_t)row * 1536 + col] = (_Float16)(oacc[tn][v] * inv);
    }
  }
}

// ---------------------------------------------------------------------------
// launch
// ---------------------------------------------------------------------------
extern "C" void kernel_launch(void* const* d_in, const int* in_sizes, int n_in,
                              void* d_out, int out_size, void* d_ws,
                              size_t ws_size, hipStream_t stream) {
  const float* x      = (const float*)d_in[0];
  const float* fore_x = (const float*)d_in[1];
  const float* post_x = (const float*)d_in[2];
  const float* Wq     = (const float*)d_in[3];
  const float* bq     = (const float*)d_in[4];
  const float* Wkv1   = (const float*)d_in[5];
  const float* bkv1   = (const float*)d_in[6];
  const float* Wkv2   = (const float*)d_in[7];
  const float* bkv2   = (const float*)d_in[8];
  const float* Wp     = (const float*)d_in[9];
  const float* bp     = (const float*)d_in[10];
  float* out          = (float*)d_out;

  // workspace carve-up (f16 elements)
  _Float16* ws = (_Float16*)d_ws;
  size_t o = 0;
  _Float16* fore_h = ws + o; o += (size_t)8 * 2048 * 256;   // 4M
  _Float16* post_h = ws + o; o += (size_t)8 * 512 * 256;    // 1M
  _Float16* Wq_h   = ws + o; o += (size_t)512 * 512;
  _Float16* Wkv1_h = ws + o; o += (size_t)256 * 1024;
  _Float16* Wkv2_h = ws + o; o += (size_t)256 * 1024;
  _Float16* Wp_h   = ws + o; o += (size_t)1536 * 512;
  _Float16* q_h    = ws + o; o += (size_t)8192 * 512;       // scaled q
  _Float16* kv1_h  = ws + o; o += (size_t)8 * 2048 * 1024;
  _Float16* kv2_h  = ws + o; o += (size_t)8 * 512 * 1024;
  _Float16* cat_h  = ws + o; o += (size_t)8192 * 1536;      // [x | x1 | x2]

  auto cvt = [&](const float* src, _Float16* dst, int n) {
    cvt_f32_f16<<<(n + 255) / 256, 256, 0, stream>>>(src, dst, n);
  };
  cvt(fore_x, fore_h, 8 * 2048 * 256);
  cvt(post_x, post_h, 8 * 512 * 256);
  cvt(Wq, Wq_h, 512 * 512);
  cvt(Wkv1, Wkv1_h, 256 * 1024);
  cvt(Wkv2, Wkv2_h, 256 * 1024);
  cvt(Wp, Wp_h, 1536 * 512);
  cvt_x_to_cat<<<(8192 * 512 + 255) / 256, 256, 0, stream>>>(x, cat_h,
                                                             8192 * 512);

  dim3 blk(256);
  // q = (x @ Wq + bq) * hd^-0.5   (A read from cat buffer, lda=1536)
  gemm_wmma<true><<<dim3(512 / 128, 8192 / 128), blk, 0, stream>>>(
      cat_h, Wq_h, bq, q_h, 8192, 512, 512, 1536, 0.125f);
  // kv1 = fore_x @ Wkv1 + bkv1
  gemm_wmma<true><<<dim3(1024 / 128, 16384 / 128), blk, 0, stream>>>(
      fore_h, Wkv1_h, bkv1, kv1_h, 16384, 1024, 256, 256, 1.0f);
  // kv2 = post_x @ Wkv2 + bkv2
  gemm_wmma<true><<<dim3(1024 / 128, 4096 / 128), blk, 0, stream>>>(
      post_h, Wkv2_h, bkv2, kv2_h, 4096, 1024, 256, 256, 1.0f);

  // x_1 -> cat[:, 512:1024),  x_2 -> cat[:, 1024:1536)
  attn_wmma<<<dim3(16, HEADS, 8), dim3(128), 0, stream>>>(q_h, kv1_h, cat_h,
                                                          2048, 512);
  attn_wmma<<<dim3(16, HEADS, 8), dim3(128), 0, stream>>>(q_h, kv2_h, cat_h,
                                                          512, 1024);

  // out = cat @ Wp + bp  (f32 out)
  gemm_wmma<false><<<dim3(512 / 128, 8192 / 128), blk, 0, stream>>>(
      cat_h, Wp_h, bp, out, 8192, 512, 1536, 1536, 1.0f);
}
